// MM_TCN_GCN_unit_1864015806764
// MI455X (gfx1250) — matrix operand order repaired and checked by reading, not compiled, not run on hardware
//
#include <hip/hip_runtime.h>
#include <hip/hip_bf16.h>
#include <math.h>

// ---------------------------------------------------------------------------
// MM-TCN-GCN unit for MI455X (gfx1250): bf16 WMMA (f32 accum) for every
// channel-mixing GEMM + attention apply.  Operand tiles staged through LDS via
// the CDNA5 async global->LDS path (ASYNCcnt); geometry fully templated so all
// tile index math constant-folds.
// ---------------------------------------------------------------------------

typedef __attribute__((ext_vector_type(16))) __bf16 v16bf;
typedef __attribute__((ext_vector_type(8)))  float  v8f;

#define NB    128
#define CCH   64
#define TTT   64
#define VVV   25
#define NCOLS 1600          // TTT*VVV
#define FELEM 13107200      // NB*CCH*TTT*VVV

// K index inside a 16x16x32 bf16 WMMA A operand for element e (0..15) of the
// per-lane v16bf, given khalf = lane>>4.  (ISA 7.12.2 "16-bit A-Matrix 16x32")
__device__ __forceinline__ constexpr int wmma_a_k(int e, int khalf) {
  int j = e >> 1, p = e & 1;
  return ((j < 4) ? 0 : 16) + (khalf << 3) + ((j & 3) << 1) + p;
}
// B operand: same K order as A (wmma_a_k), column = lane&15.  C/D: row = i + 8*khalf.

// Async copy one f32 from global memory into LDS (byte offset), ASYNCcnt-tracked.
__device__ __forceinline__ void async_f32_to_lds(const float* g, unsigned ldsOff) {
  asm volatile("global_load_async_to_lds_b32 %0, %1, off"
               :: "v"(ldsOff), "v"(g) : "memory");
}
__device__ __forceinline__ void wait_async0() {
  asm volatile("s_wait_asynccnt 0x0" ::: "memory");
}

// ---------------------------------------------------------------------------
// Generic 1x1-conv GEMM:  out[n, och_off+m, col] (=|+=) epi( sum_c W[m,c]*X[n,c,col] )
// X channels: c < CA -> xA, else xB (concat support).  MODE: 0=bias, 1=bias+bn+relu,
// 2=bias+bn, 3=bias+bn+res+relu.
// ---------------------------------------------------------------------------
template<int O, int Cin, int CA, int MODE>
__global__ __launch_bounds__(256) void k_conv1x1(
    float* __restrict__ out, int oc_total, int och_off,
    const float* __restrict__ xA, const float* __restrict__ xB,
    const float* __restrict__ Wm, const float* __restrict__ wb,
    const float* __restrict__ bns, const float* __restrict__ bnb,
    const float* __restrict__ res, int accum)
{
  constexpr int rowTiles   = O / 16;                 // 1 (O=16) or 4 (O=64)
  constexpr int colsPB     = (8 / rowTiles) * 16;    // 128 or 32
  constexpr int blocksPerN = (NCOLS + colsPB - 1) / colsPB;
  constexpr int OP   = O + 1;                        // padded k-major weight stride
  constexpr int BSTR = colsPB + 4;                   // padded B-tile stride
  constexpr bool exactCols = (NCOLS % colsPB) == 0;

  __shared__ float ldsW[Cin * OP];
  __shared__ float ldsB[32 * BSTR];

  int n  = blockIdx.x / blocksPerN;
  int cb = blockIdx.x % blocksPerN;

  // Stage weights once (transposed to k-major) and zero the B tile (covers
  // out-of-range columns so WMMA gathers need no guards).
  for (int idx = threadIdx.x; idx < O * Cin; idx += 256)
    ldsW[(idx % Cin) * OP + (idx / Cin)] = Wm[idx];
  for (int i = threadIdx.x; i < 32 * BSTR; i += 256) ldsB[i] = 0.f;

  int wave = threadIdx.x >> 5, lane = threadIdx.x & 31;
  int rt = wave % rowTiles, ct = wave / rowTiles;
  int col0b = cb * colsPB;
  int col   = col0b + (ct << 4) + (lane & 15);
  int khalf = lane >> 4;
  int mrow  = (rt << 4) + (lane & 15);
  bool fullBlk = exactCols || (col0b + colsPB <= NCOLS);

  v8f acc = {};
  for (int kc = 0; kc < Cin; kc += 32) {
    __syncthreads();   // prior chunk consumed (1st iter: staging/zero visible)
    // Async-stage 32 x colsPB activation tile into LDS.
    if (fullBlk) {
#pragma unroll
      for (int s = 0; s < 32 * colsPB / 256; ++s) {
        int i = threadIdx.x + s * 256;
        int kb = i / colsPB, cc = i % colsPB;        // compile-time shifts/masks
        int c  = kc + kb;
        const float* src = (Cin == CA || c < CA)
            ? (xA + ((size_t)n * CA + c) * NCOLS)
            : (xB + ((size_t)n * CA + (c - CA)) * NCOLS);
        async_f32_to_lds(src + col0b + cc, (unsigned)(size_t)(&ldsB[kb * BSTR + cc]));
      }
    } else {
#pragma unroll
      for (int s = 0; s < 32 * colsPB / 256; ++s) {
        int i = threadIdx.x + s * 256;
        int kb = i / colsPB, cc = i % colsPB;
        int c  = kc + kb;
        const float* src = (Cin == CA || c < CA)
            ? (xA + ((size_t)n * CA + c) * NCOLS)
            : (xB + ((size_t)n * CA + (c - CA)) * NCOLS);
        if (col0b + cc < NCOLS)
          async_f32_to_lds(src + col0b + cc, (unsigned)(size_t)(&ldsB[kb * BSTR + cc]));
      }
    }
    wait_async0();
    __syncthreads();

    v16bf av, bv;
#pragma unroll
    for (int e = 0; e < 16; ++e) {
      av[e] = (__bf16)ldsW[(kc + wmma_a_k(e, khalf)) * OP + mrow];
      bv[e] = (__bf16)ldsB[wmma_a_k(e, khalf) * BSTR + (ct << 4) + (lane & 15)];
    }
    acc = __builtin_amdgcn_wmma_f32_16x16x32_bf16(false, av, false, bv,
                                                  (short)0, acc, false, false);
  }

  if (exactCols || col < NCOLS) {
#pragma unroll
    for (int i = 0; i < 8; ++i) {
      int m = (rt << 4) + i + (khalf << 3);          // always < O
      float v = acc[i] + wb[m];
      if (MODE >= 1) v = v * bns[m] + bnb[m];
      size_t oidx = ((size_t)n * oc_total + och_off + m) * NCOLS + col;
      if (MODE == 3) v += res[oidx];
      if (MODE == 1 || MODE == 3) v = fmaxf(v, 0.f);
      if (accum) out[oidx] += v; else out[oidx] = v;
    }
  }
}

// ---------------------------------------------------------------------------
// Attention apply: block per (n,o).  Build a[u][v] = wb[o] + sum_r w[o,r]*rel[n,r,u,v]
// (+PA[u,v]) (+alpha) into a zero-padded LDS tile, async-stage v[n,o] (strided so
// gcn2's transposed geometry reuses standard-layout v) into another zero-padded
// LDS tile, then y[n,o,t,u] += sum_v v[t,v]*a[u,v] via branchless WMMA.
// ---------------------------------------------------------------------------
template<int T, int J, int ST, int SK>
__global__ __launch_bounds__(256) void k_attn(
    float* __restrict__ yacc, const float* __restrict__ vmat,
    const float* __restrict__ rel,
    const float* __restrict__ w, const float* __restrict__ wb,
    const float* __restrict__ PA, const float* __restrict__ alpha)
{
  constexpr int rowTiles = (T + 15) / 16;            // 4 (T=64) or 2 (T=25)
  constexpr int TP = rowTiles * 16;                  // 64 or 32
  constexpr int colTiles = 8 / rowTiles;             // 2 or 4
  constexpr int UP = colTiles * 16;                  // 32 or 64
  constexpr int JP2 = ((J + 31) / 32) * 32;          // 32 or 64
  constexpr int JPA = JP2 + 1;

  __shared__ float amat[UP * JPA];
  __shared__ float ldsV[TP * JPA];

  int n = blockIdx.x >> 6;
  int o = blockIdx.x & 63;

  for (int i = threadIdx.x; i < UP * JPA; i += 256) amat[i] = 0.f;
  for (int i = threadIdx.x; i < TP * JPA; i += 256) ldsV[i] = 0.f;
  __syncthreads();

  const float* rb = rel + (size_t)n * 8 * J * J;
  for (int idx = threadIdx.x; idx < J * J; idx += 256) {
    int u = idx / J, vv = idx % J;
    float s = wb[o];
#pragma unroll
    for (int r = 0; r < 8; ++r)
      s += w[o * 8 + r] * rb[r * J * J + u * J + vv];
    if (PA)    s += PA[u * J + vv];
    if (alpha) s += alpha[0];
    amat[u * JPA + vv] = s;
  }
  const float* vbase = vmat + (size_t)(n * 64 + o) * T * J;
  for (int idx = threadIdx.x; idx < T * J; idx += 256) {
    int t = idx / J, v = idx % J;
    async_f32_to_lds(vbase + t * ST + v * SK, (unsigned)(size_t)(&ldsV[t * JPA + v]));
  }
  wait_async0();
  __syncthreads();

  int wave = threadIdx.x >> 5, lane = threadIdx.x & 31;
  int rt = wave % rowTiles, ct = wave / rowTiles;
  int khalf = lane >> 4;
  int trow = (rt << 4) + (lane & 15);
  int col  = (ct << 4) + (lane & 15);                // u index

  v8f acc = {};
#pragma unroll
  for (int kc = 0; kc < J; kc += 32) {
    v16bf av, bv;
#pragma unroll
    for (int e = 0; e < 16; ++e) {
      av[e] = (__bf16)ldsV[trow * JPA + kc + wmma_a_k(e, khalf)];
      bv[e] = (__bf16)amat[col * JPA + kc + wmma_a_k(e, khalf)];
    }
    acc = __builtin_amdgcn_wmma_f32_16x16x32_bf16(false, av, false, bv,
                                                  (short)0, acc, false, false);
  }
#pragma unroll
  for (int i = 0; i < 8; ++i) {
    int t = (rt << 4) + i + (khalf << 3);
    if (t < T && col < J)
      yacc[((size_t)(n * 64 + o) * T + t) * J + col] += acc[i];
  }
}

// ---------------------------------------------------------------------------
// Small scalar kernels
// ---------------------------------------------------------------------------
__global__ void k_mean(float* __restrict__ xm, const float* __restrict__ x,
                       int J, int L, int strideJ, int strideL) {
  int idx = blockIdx.x * blockDim.x + threadIdx.x;
  if (idx >= NB * CCH * J) return;
  int jj = idx % J; int nc = idx / J;
  const float* base = x + (size_t)nc * NCOLS;
  float s = 0.f;
  for (int l = 0; l < L; ++l) s += base[jj * strideJ + l * strideL];
  xm[idx] = s / (float)L;
}

__global__ void k_qk(float* __restrict__ q, const float* __restrict__ xm,
                     const float* __restrict__ w, const float* __restrict__ b, int J) {
  int idx = blockIdx.x * blockDim.x + threadIdx.x;
  if (idx >= NB * 8 * J) return;
  int u = idx % J; int r = (idx / J) % 8; int n = idx / (J * 8);
  float s = b[r];
  for (int c = 0; c < CCH; ++c) s += w[r * CCH + c] * xm[((size_t)n * CCH + c) * J + u];
  q[idx] = s;
}

__global__ void k_rel(float* __restrict__ rel, const float* __restrict__ q,
                      const float* __restrict__ k, int J) {
  int idx = blockIdx.x * blockDim.x + threadIdx.x;
  if (idx >= NB * 8 * J * J) return;
  int v = idx % J; int u = (idx / J) % J; int nr = idx / (J * J);
  rel[idx] = tanhf(q[nr * J + u] - k[nr * J + v]);
}

// y = relu(bn(ysum) + xres); optional transpose of the last two dims on output.
__global__ void k_gcn_post(float* __restrict__ out, const float* __restrict__ ysum,
                           const float* __restrict__ s, const float* __restrict__ b,
                           const float* __restrict__ xres, int A, int B_, int transpose) {
  int idx = blockIdx.x * blockDim.x + threadIdx.x;
  if (idx >= NB * CCH * A * B_) return;
  int bb = idx % B_; int aa = (idx / B_) % A;
  int c = (idx / (A * B_)) % CCH; int n = idx / (CCH * A * B_);
  size_t oidx = transpose ? (((size_t)(n * CCH + c) * B_ + bb) * A + aa) : (size_t)idx;
  float v = ysum[idx] * s[c] + b[c] + xres[oidx];
  out[oidx] = fmaxf(v, 0.f);
}

// Dilated temporal conv (k=5, pad=2*dil) + BN, into out channel slot choff.
__global__ void k_tconv(float* __restrict__ out, int choff, const float* __restrict__ h,
                        const float* __restrict__ tw, const float* __restrict__ tb,
                        const float* __restrict__ s2, const float* __restrict__ b2,
                        int dil, int accum) {
  int idx = blockIdx.x * blockDim.x + threadIdx.x;
  if (idx >= NB * 16 * NCOLS) return;
  int v = idx % VVV; int t = (idx / VVV) % TTT;
  int o = (idx / NCOLS) % 16; int n = idx / (16 * NCOLS);
  float acc = tb[o];
  for (int i = 0; i < 16; ++i) {
    const float* hb = h + ((size_t)(n * 16 + i) * TTT) * VVV + v;
#pragma unroll
    for (int kk = 0; kk < 5; ++kk) {
      int tt = t + (kk - 2) * dil;
      if (tt >= 0 && tt < TTT) acc += tw[(o * 16 + i) * 5 + kk] * hb[tt * VVV];
    }
  }
  float val = acc * s2[o] + b2[o];
  size_t oidx = ((size_t)(n * CCH + choff + o) * TTT + t) * VVV + v;
  if (accum) out[oidx] += val; else out[oidx] = val;
}

// MaxPool(3 over T, pad 1 with -inf) + BN into channel slot choff.
__global__ void k_mpbn(float* __restrict__ out, int choff, const float* __restrict__ h,
                       const float* __restrict__ s2, const float* __restrict__ b2, int accum) {
  int idx = blockIdx.x * blockDim.x + threadIdx.x;
  if (idx >= NB * 16 * NCOLS) return;
  int v = idx % VVV; int t = (idx / VVV) % TTT;
  int o = (idx / NCOLS) % 16; int n = idx / (16 * NCOLS);
  const float* hb = h + ((size_t)(n * 16 + o) * TTT) * VVV + v;
  float m = -3.4e38f;
#pragma unroll
  for (int dt = -1; dt <= 1; ++dt) {
    int tt = t + dt;
    if (tt >= 0 && tt < TTT) m = fmaxf(m, hb[tt * VVV]);
  }
  float val = m * s2[o] + b2[o];
  size_t oidx = ((size_t)(n * CCH + choff + o) * TTT + t) * VVV + v;
  if (accum) out[oidx] += val; else out[oidx] = val;
}

__global__ void k_addrelu(float* __restrict__ y, const float* __restrict__ xres, int total) {
  int idx = blockIdx.x * blockDim.x + threadIdx.x;
  if (idx < total) y[idx] = fmaxf(y[idx] + xres[idx], 0.f);
}

// ---------------------------------------------------------------------------
// Host orchestration
// ---------------------------------------------------------------------------
extern "C" void kernel_launch(void* const* d_in, const int* in_sizes, int n_in,
                              void* d_out, int out_size, void* d_ws, size_t ws_size,
                              hipStream_t stream) {
  (void)in_sizes; (void)n_in; (void)out_size; (void)ws_size;
  auto in = [&](int i) { return (const float*)d_in[i]; };
  const float* x1 = in(0);
  const float* x2 = in(1);

  const size_t F = FELEM;
  float* W   = (float*)d_ws;
  float* X   = W;                // conv1 output
  float* YS1 = W + 1 * F;        // gcn sums / tcn accum A
  float* YS2 = W + 2 * F;
  float* V1  = W + 3 * F;        // v-proj / m12t
  float* V2  = W + 4 * F;
  float* M11 = W + 5 * F;
  float* M21 = W + 6 * F;
  float* H   = W + 7 * F;                         // 3,276,800
  float* XM1 = H + (size_t)NB * 16 * NCOLS;       // 524,288
  float* XM2 = XM1 + (size_t)NB * CCH * TTT;
  float* QB  = XM2 + (size_t)NB * CCH * TTT;      // 65,536
  float* KB  = QB + (size_t)NB * 8 * TTT;
  float* REL = KB + (size_t)NB * 8 * TTT;         // up to 4,194,304

  auto cdiv = [](int a, int b) { return (a + b - 1) / b; };
  const int G64 = NB * 50;   // O=64: colsPB=32 -> 50 blocks per sample
  const int G16 = NB * 13;   // O=16: colsPB=128 -> 13 blocks per sample

  // ---- x = relu(bn1(conv1(cat(x1,x2)))) ----
  k_conv1x1<64,128,64,1><<<G64, 256, 0, stream>>>(X, 64, 0, x1, x2,
      in(2), in(3), in(4), in(5), nullptr, 0);

  // ---- gcn1 (3 subsets, PA, standard geometry T=64, J=25) ----
  hipMemsetAsync(YS1, 0, F * sizeof(float), stream);
  hipMemsetAsync(YS2, 0, F * sizeof(float), stream);
  k_mean<<<cdiv(NB * CCH * 25, 256), 256, 0, stream>>>(XM1, x1, 25, 64, 1, 25);
  k_mean<<<cdiv(NB * CCH * 25, 256), 256, 0, stream>>>(XM2, x2, 25, 64, 1, 25);
  for (int sb = 0; sb < 3; ++sb) {
    int bs = 10 + sb * 12;
    k_qk<<<cdiv(NB * 8 * 25, 256), 256, 0, stream>>>(QB, XM1, in(bs + 0), in(bs + 1), 25);
    k_qk<<<cdiv(NB * 8 * 25, 256), 256, 0, stream>>>(KB, XM2, in(bs + 2), in(bs + 3), 25);
    k_rel<<<cdiv(NB * 8 * 625, 256), 256, 0, stream>>>(REL, QB, KB, 25);
    k_conv1x1<64,64,64,0><<<G64, 256, 0, stream>>>(V1, 64, 0, x1, x1,
        in(bs + 4), in(bs + 5), nullptr, nullptr, nullptr, 0);
    k_conv1x1<64,64,64,0><<<G64, 256, 0, stream>>>(V2, 64, 0, x2, x2,
        in(bs + 6), in(bs + 7), nullptr, nullptr, nullptr, 0);
    k_attn<64,25,25,1><<<NB * 64, 256, 0, stream>>>(YS1, V1, REL,
        in(bs + 8), in(bs + 9), in(51) + sb * 625, nullptr);
    k_attn<64,25,25,1><<<NB * 64, 256, 0, stream>>>(YS2, V2, REL,
        in(bs + 10), in(bs + 11), nullptr, in(46));
  }
  k_gcn_post<<<cdiv((int)F, 256), 256, 0, stream>>>(M11, YS1, in(47), in(48), x1, 64, 25, 0);
  k_gcn_post<<<cdiv((int)F, 256), 256, 0, stream>>>(M21, YS2, in(49), in(50), x2, 64, 25, 0);

  // ---- gcn2 (1 subset, transposed geometry T=25, J=64) ----
  hipMemsetAsync(YS1, 0, F * sizeof(float), stream);
  hipMemsetAsync(YS2, 0, F * sizeof(float), stream);
  k_mean<<<cdiv(NB * CCH * 64, 256), 256, 0, stream>>>(XM1, x1, 64, 25, 25, 1);
  k_mean<<<cdiv(NB * CCH * 64, 256), 256, 0, stream>>>(XM2, x2, 64, 25, 25, 1);
  k_qk<<<cdiv(NB * 8 * 64, 256), 256, 0, stream>>>(QB, XM1, in(52), in(53), 64);
  k_qk<<<cdiv(NB * 8 * 64, 256), 256, 0, stream>>>(KB, XM2, in(54), in(55), 64);
  k_rel<<<cdiv(NB * 8 * 4096, 256), 256, 0, stream>>>(REL, QB, KB, 64);
  k_conv1x1<64,64,64,0><<<G64, 256, 0, stream>>>(V1, 64, 0, x1, x1,
      in(56), in(57), nullptr, nullptr, nullptr, 0);
  k_conv1x1<64,64,64,0><<<G64, 256, 0, stream>>>(V2, 64, 0, x2, x2,
      in(58), in(59), nullptr, nullptr, nullptr, 0);
  k_attn<25,64,1,25><<<NB * 64, 256, 0, stream>>>(YS1, V1, REL,
      in(60), in(61), nullptr, nullptr);
  k_attn<25,64,1,25><<<NB * 64, 256, 0, stream>>>(YS2, V2, REL,
      in(62), in(63), nullptr, in(64));
  // write back in standard layout (transpose folded in): m12t -> V1, m22t -> V2
  k_gcn_post<<<cdiv((int)F, 256), 256, 0, stream>>>(V1, YS1, in(65), in(66), x1, 25, 64, 1);
  k_gcn_post<<<cdiv((int)F, 256), 256, 0, stream>>>(V2, YS2, in(67), in(68), x2, 25, 64, 1);

  // ---- tcn (4 invocations: accumulate pairs into YS1 / YS2) ----
  auto run_tcn = [&](const float* xin, float* outb, int accum) {
    k_conv1x1<16,64,64,1><<<G16, 256, 0, stream>>>(H, 16, 0, xin, xin,
        in(69), in(70), in(71), in(72), nullptr, 0);
    k_tconv<<<cdiv(NB * 16 * NCOLS, 256), 256, 0, stream>>>(outb, 0, H, in(73), in(74),
                                                            in(75), in(76), 1, accum);
    k_conv1x1<16,64,64,1><<<G16, 256, 0, stream>>>(H, 16, 0, xin, xin,
        in(77), in(78), in(79), in(80), nullptr, 0);
    k_tconv<<<cdiv(NB * 16 * NCOLS, 256), 256, 0, stream>>>(outb, 16, H, in(81), in(82),
                                                            in(83), in(84), 2, accum);
    k_conv1x1<16,64,64,1><<<G16, 256, 0, stream>>>(H, 16, 0, xin, xin,
        in(85), in(86), in(87), in(88), nullptr, 0);
    k_mpbn<<<cdiv(NB * 16 * NCOLS, 256), 256, 0, stream>>>(outb, 32, H, in(89), in(90), accum);
    k_conv1x1<16,64,64,2><<<G16, 256, 0, stream>>>(outb, 64, 48, xin, xin,
        in(91), in(92), in(93), in(94), nullptr, accum);
  };
  run_tcn(M11, YS1, 0);
  run_tcn(V1,  YS1, 1);   // tcn(m12t)
  run_tcn(M21, YS2, 0);
  run_tcn(V2,  YS2, 1);   // tcn(m22t)

  // ---- y1 = relu(tcn_sum1 + x1); y2 = relu(tcn_sum2 + x2) ----
  k_addrelu<<<cdiv((int)F, 256), 256, 0, stream>>>(YS1, x1, (int)F);
  k_addrelu<<<cdiv((int)F, 256), 256, 0, stream>>>(YS2, x2, (int)F);

  // ---- out = relu(x + bn2(conv2(cat(y1,y2)))) ----
  k_conv1x1<64,128,64,3><<<G64, 256, 0, stream>>>((float*)d_out, 64, 0, YS1, YS2,
      in(6), in(7), in(8), in(9), X, 0);
}
